// Model_70291434766564
// MI455X (gfx1250) — matrix operand order repaired and checked by reading, not compiled, not run on hardware
//
#include <hip/hip_runtime.h>
#include <hip/hip_bf16.h>
#include <cmath>

typedef _Float16 half8 __attribute__((ext_vector_type(8)));
typedef _Float16 v16h  __attribute__((ext_vector_type(16)));
typedef float    v8f   __attribute__((ext_vector_type(8)));

// ---------------------------------------------------------------- utilities
__global__ void k_zero(float* p, size_t n) {
    size_t i = (size_t)blockIdx.x * blockDim.x + threadIdx.x;
    size_t st = (size_t)gridDim.x * blockDim.x;
    for (; i < n; i += st) p[i] = 0.0f;
}

__global__ void k_cvt_f16(const float* __restrict__ in, _Float16* __restrict__ out, size_t n) {
    size_t i = (size_t)blockIdx.x * blockDim.x + threadIdx.x;
    size_t st = (size_t)gridDim.x * blockDim.x;
    for (; i < n; i += st) out[i] = (_Float16)in[i];
}

// copy embeddings (N x 128) into output (N x 256) columns [0,128)
__global__ void k_emb_copy(const float* __restrict__ emb, float* __restrict__ out, size_t total) {
    size_t i = (size_t)blockIdx.x * blockDim.x + threadIdx.x;
    size_t st = (size_t)gridDim.x * blockDim.x;
    for (; i < total; i += st) {
        size_t r = i >> 7;        // /128
        size_t c = i & 127;
        out[r * 256 + c] = emb[i];
    }
}

// ---------------------------------------------------------------- edge logits
__global__ void k_edge_logits(const float* __restrict__ hsrc, const float* __restrict__ hdst,
                              int D, const int* __restrict__ eu, const int* __restrict__ ei,
                              float* __restrict__ logit, int E, float scale) {
    int e = blockIdx.x * blockDim.x + threadIdx.x;
    if (e >= E) return;
    const float4* a = (const float4*)(hsrc + (size_t)eu[e] * D);
    const float4* b = (const float4*)(hdst + (size_t)ei[e] * D);
    float s = 0.0f;
    int n4 = D >> 2;
    for (int d = 0; d < n4; ++d) {
        float4 x = a[d], y = b[d];
        s += x.x * y.x + x.y * y.y + x.z * y.z + x.w * y.w;
    }
    logit[e] = s * scale;
}

// ---------------------------------------------------------------- global softmax
__global__ void k_block_max(const float* __restrict__ x, int n, float* __restrict__ part) {
    __shared__ float s[256];
    float m = -INFINITY;
    for (int i = blockIdx.x * blockDim.x + threadIdx.x; i < n; i += gridDim.x * blockDim.x)
        m = fmaxf(m, x[i]);
    s[threadIdx.x] = m;
    __syncthreads();
    for (int st = 128; st > 0; st >>= 1) {
        if (threadIdx.x < st) s[threadIdx.x] = fmaxf(s[threadIdx.x], s[threadIdx.x + st]);
        __syncthreads();
    }
    if (threadIdx.x == 0) part[blockIdx.x] = s[0];
}

__global__ void k_final_max(const float* __restrict__ part, int n, float* __restrict__ out) {
    __shared__ float s[256];
    float m = -INFINITY;
    for (int i = threadIdx.x; i < n; i += 256) m = fmaxf(m, part[i]);
    s[threadIdx.x] = m;
    __syncthreads();
    for (int st = 128; st > 0; st >>= 1) {
        if (threadIdx.x < st) s[threadIdx.x] = fmaxf(s[threadIdx.x], s[threadIdx.x + st]);
        __syncthreads();
    }
    if (threadIdx.x == 0) out[0] = s[0];
}

__global__ void k_block_sumexp(const float* __restrict__ x, int n, const float* __restrict__ mx,
                               float* __restrict__ part) {
    __shared__ float s[256];
    float m = mx[0];
    float a = 0.0f;
    for (int i = blockIdx.x * blockDim.x + threadIdx.x; i < n; i += gridDim.x * blockDim.x)
        a += __expf(x[i] - m);
    s[threadIdx.x] = a;
    __syncthreads();
    for (int st = 128; st > 0; st >>= 1) {
        if (threadIdx.x < st) s[threadIdx.x] += s[threadIdx.x + st];
        __syncthreads();
    }
    if (threadIdx.x == 0) part[blockIdx.x] = s[0];
}

__global__ void k_final_sum(const float* __restrict__ part, int n, float* __restrict__ out) {
    __shared__ float s[256];
    float a = 0.0f;
    for (int i = threadIdx.x; i < n; i += 256) a += part[i];
    s[threadIdx.x] = a;
    __syncthreads();
    for (int st = 128; st > 0; st >>= 1) {
        if (threadIdx.x < st) s[threadIdx.x] += s[threadIdx.x + st];
        __syncthreads();
    }
    if (threadIdx.x == 0) out[0] = s[0];
}

__global__ void k_softmax_norm(float* __restrict__ x, int n, const float* __restrict__ scal) {
    int i = blockIdx.x * blockDim.x + threadIdx.x;
    if (i >= n) return;
    x[i] = __expf(x[i] - scal[0]) / scal[1];
}

// ---------------------------------------------------------------- WMMA GEMM + bias + relu
// out(R x 64) = relu( A(R x K, f16) * B(64 x K, f16)^T + bias )
// One wave computes a 16x64 tile: 4 n-tiles, shared A fragment.
__global__ void k_gemm_relu_wmma(const _Float16* __restrict__ A, const _Float16* __restrict__ Bm,
                                 const float* __restrict__ bias, float* __restrict__ out,
                                 int R, int K) {
    int gid  = blockIdx.x * blockDim.x + threadIdx.x;
    int wave = gid >> 5;
    int lane = threadIdx.x & 31;
    if (wave * 16 >= R) return;           // whole-wave guard: EXEC all-ones inside
    int hk = lane >> 4;                    // K-half select
    int r  = lane & 15;                    // M (for A) / N (for B) index

    v8f acc[4] = {};
    for (int kb = 0; kb < K; kb += 32) {
        // A fragment: lane<16 -> M=r, K=kb..kb+15; lanes>=16 -> K=kb+16..kb+31
        const _Float16* ap = A + (size_t)(wave * 16 + r) * K + kb + hk * 16;
        half8 lo = *(const half8*)ap;
        half8 hi = *(const half8*)(ap + 8);
        v16h a = __builtin_shufflevector(lo, hi, 0, 1, 2, 3, 4, 5, 6, 7,
                                                 8, 9, 10, 11, 12, 13, 14, 15);
#pragma unroll
        for (int nt = 0; nt < 4; ++nt) {
            // B fragment: lane<16 -> N=r, K=kb..kb+15; lanes>=16 -> K=kb+16..kb+31
            const _Float16* bp = Bm + (size_t)(nt * 16 + r) * K + kb + hk * 16;
            v16h b = *(const v16h*)bp;
            acc[nt] = __builtin_amdgcn_wmma_f32_16x16x32_f16(
                false, a, false, b, (short)0, acc[nt], false, false);
        }
    }
#pragma unroll
    for (int nt = 0; nt < 4; ++nt) {
#pragma unroll
        for (int v = 0; v < 8; ++v) {
            int row = wave * 16 + v + 8 * hk;   // C/D layout: M = v + 8*(lane/16)
            int col = nt * 16 + r;              // N = lane & 15
            float x = acc[nt][v] + bias[col];
            out[(size_t)row * 64 + col] = fmaxf(x, 0.0f);
        }
    }
}

// ---------------------------------------------------------------- alpha-weighted scatter (both directions)
__global__ void k_scatter(const int* __restrict__ eu, const int* __restrict__ ei,
                          const float* __restrict__ alpha,
                          const float* __restrict__ fsrc, const float* __restrict__ fdst,
                          float* __restrict__ user_new, float* __restrict__ item_new, int E) {
    size_t id = (size_t)blockIdx.x * blockDim.x + threadIdx.x;
    int e = (int)(id >> 6);
    int c = (int)(id & 63);
    if (e >= E) return;
    float a = alpha[e];
    int u = eu[e], i = ei[e];
    atomicAdd(&item_new[(size_t)i * 64 + c], fsrc[(size_t)u * 64 + c] * a);
    atomicAdd(&user_new[(size_t)u * 64 + c], fdst[(size_t)i * 64 + c] * a);
}

// ---------------------------------------------------------------- row l2-norm into output slice
__global__ void k_l2norm_out(const float* __restrict__ h, float* __restrict__ out,
                             int ncols_out, int col_off) {
    int row = blockIdx.x;
    int c = threadIdx.x;                   // 64 threads
    float v = h[(size_t)row * 64 + c];
    __shared__ float s[64];
    s[c] = v * v;
    __syncthreads();
    for (int st = 32; st > 0; st >>= 1) {
        if (c < st) s[c] += s[c + st];
        __syncthreads();
    }
    float n = fmaxf(sqrtf(s[0]), 1e-12f);
    out[(size_t)row * ncols_out + col_off + c] = v / n;
}

// ---------------------------------------------------------------- launch
extern "C" void kernel_launch(void* const* d_in, const int* in_sizes, int n_in,
                              void* d_out, int out_size, void* d_ws, size_t ws_size,
                              hipStream_t stream) {
    const float* emb = (const float*)d_in[0];
    const float* W1s = (const float*)d_in[1];
    const float* b1s = (const float*)d_in[2];
    const float* W1d = (const float*)d_in[3];
    const float* b1d = (const float*)d_in[4];
    const float* W2s = (const float*)d_in[5];
    const float* b2s = (const float*)d_in[6];
    const float* W2d = (const float*)d_in[7];
    const float* b2d = (const float*)d_in[8];
    const int* eu = (const int*)d_in[9];
    const int* ei = (const int*)d_in[10];

    const int E  = in_sizes[9];
    const int NU = 60000, NI = 40000;      // fixed by setup_inputs()
    const int N  = NU + NI;
    const int D0 = in_sizes[0] / N;        // 128
    float* outp = (float*)d_out;

    // workspace carve-up (float-granular, 256B aligned regions)
    float* W = (float*)d_ws;
    size_t off = 0;
    auto alloc = [&](size_t nf) { size_t o = off; off += (nf + 63) & ~(size_t)63; return o; };
    float* alpha = W + alloc((size_t)E);
    float* part  = W + alloc(1024 + 64);
    float* scal  = part + 1024;            // [0]=max, [1]=sum
    float* h1    = W + alloc((size_t)N * 64);
    float* h2    = W + alloc((size_t)N * 64);
    float* fsrc  = W + alloc((size_t)NU * 64);
    float* fdst  = W + alloc((size_t)NI * 64);
    _Float16* h16  = (_Float16*)(W + alloc((size_t)N * D0 / 2 + 64));
    _Float16* w16s = (_Float16*)(W + alloc((size_t)64 * D0 / 2 + 64));
    _Float16* w16d = (_Float16*)(W + alloc((size_t)64 * D0 / 2 + 64));

    auto cdiv = [](size_t a, size_t b) { return (unsigned)((a + b - 1) / b); };

    auto run_layer = [&](const float* h, int D, const float* Ws, const float* bs,
                         const float* Wd, const float* bd, float* hout) {
        const float* hs = h;
        const float* hd = h + (size_t)NU * D;
        float scale = 1.0f / sqrtf((float)D);

        k_edge_logits<<<cdiv(E, 256), 256, 0, stream>>>(hs, hd, D, eu, ei, alpha, E, scale);
        k_block_max<<<1024, 256, 0, stream>>>(alpha, E, part);
        k_final_max<<<1, 256, 0, stream>>>(part, 1024, scal);
        k_block_sumexp<<<1024, 256, 0, stream>>>(alpha, E, scal, part);
        k_final_sum<<<1, 256, 0, stream>>>(part, 1024, scal + 1);
        k_softmax_norm<<<cdiv(E, 256), 256, 0, stream>>>(alpha, E, scal);

        size_t nh = (size_t)N * D;
        k_cvt_f16<<<cdiv(nh, 256), 256, 0, stream>>>(h, h16, nh);
        k_cvt_f16<<<cdiv((size_t)64 * D, 256), 256, 0, stream>>>(Ws, w16s, (size_t)64 * D);
        k_cvt_f16<<<cdiv((size_t)64 * D, 256), 256, 0, stream>>>(Wd, w16d, (size_t)64 * D);

        // feat_src = relu(h_src @ Ws^T + bs), feat_dst = relu(h_dst @ Wd^T + bd)
        k_gemm_relu_wmma<<<cdiv((size_t)(NU / 16) * 32, 256), 256, 0, stream>>>(
            h16, w16s, bs, fsrc, NU, D);
        k_gemm_relu_wmma<<<cdiv((size_t)(NI / 16) * 32, 256), 256, 0, stream>>>(
            h16 + (size_t)NU * D, w16d, bd, fdst, NI, D);

        k_zero<<<cdiv((size_t)N * 64, 256), 256, 0, stream>>>(hout, (size_t)N * 64);
        k_scatter<<<cdiv((size_t)E * 64, 256), 256, 0, stream>>>(
            eu, ei, alpha, fsrc, fdst, hout, hout + (size_t)NU * 64, E);
    };

    // cache[0] = embeddings
    k_emb_copy<<<cdiv((size_t)N * D0, 256), 256, 0, stream>>>(emb, outp, (size_t)N * D0);
    // layer 1 (input = embeddings, D=128); layer-2 input is UNNORMALIZED h1
    run_layer(emb, D0, W1s, b1s, W1d, b1d, h1);
    k_l2norm_out<<<N, 64, 0, stream>>>(h1, outp, 256, 128);
    // layer 2 (input = raw h1, D=64)
    run_layer(h1, 64, W2s, b2s, W2d, b2d, h2);
    k_l2norm_out<<<N, 64, 0, stream>>>(h2, outp, 256, 192);
}
